// LogComputeFullCell_29497835388997
// MI455X (gfx1250) — matrix operand order repaired and checked by reading, not compiled, not run on hardware
//
#include <hip/hip_runtime.h>
#include <hip/hip_bf16.h>

// ---------------------------------------------------------------------------
// Log-space gated RNN cell for MI455X (gfx1250, wave32, WMMA + TDM).
//
//  Phase 1: GEMM  [T*B,1024] x [2048,1024]^T  -> A (x@Wx^T+b), G (x@Wd^T+bd)
//           bf16 WMMA 16x16x32, fp32 accumulate. x-tile staged to LDS by the
//           Tensor Data Mover (tensor_load_to_lds) with bank-skew padding;
//           B fragments double-buffered to overlap global loads with WMMA.
//  Phase 2: persistent recurrent kernel, 64 WGs (one 16-channel tile each),
//           R_h fragments pinned in VGPRs, 1 grid barrier per timestep.
// ---------------------------------------------------------------------------

typedef __attribute__((ext_vector_type(16))) __bf16 v16bf;
typedef __attribute__((ext_vector_type(8)))  __bf16 v8bf;
typedef __attribute__((ext_vector_type(8)))  float  v8f;
typedef unsigned int __attribute__((ext_vector_type(4))) u32x4;
typedef int __attribute__((ext_vector_type(4)))          i32x4;
typedef int __attribute__((ext_vector_type(8)))          i32x8;

#define T_STEPS 2048
#define BATCH   16
#define DIM     1024

// LDS x-tile: 16 rows x 1024 bf16, TDM pad = 16B per 1024B of stream
// -> element e lives at byte  2*e + 16*(e>>9);  row stride = 2080B.
#define XTILE_LDS_BYTES (16 * 2080)

// --- WMMA fragment loaders (ISA 7.12.2 layouts, bf16, K=32) ----------------
// A 16x32: lane -> row m=lane%16; elems 0..7 = K(base..base+7), 8..15 = K(16+base..),
//          base = lane<16 ? 0 : 8
__device__ __forceinline__ v16bf frag_from_halves(v8bf lo, v8bf hi) {
  v16bf r;
#pragma unroll
  for (int i = 0; i < 8; ++i) { r[i] = lo[i]; r[i + 8] = hi[i]; }
  return r;
}

__device__ __forceinline__ v16bf load_frag_a(const __bf16* row, int k0, int lane) {
  const __bf16* p = row + k0 + ((lane < 16) ? 0 : 8);
  return frag_from_halves(*(const v8bf*)(p), *(const v8bf*)(p + 16));
}

// A fragment out of the TDM-padded LDS x-tile
__device__ __forceinline__ v16bf load_frag_a_lds(const char* sx, int row, int k0,
                                                 int lane) {
  unsigned e0 = (unsigned)(row * DIM + k0 + ((lane < 16) ? 0 : 8));
  unsigned e1 = e0 + 16;                       // same 512-elem pad interval
  v8bf lo = *(const v8bf*)(sx + 2u * e0 + ((e0 >> 9) << 4));
  v8bf hi = *(const v8bf*)(sx + 2u * e1 + ((e1 >> 9) << 4));
  return frag_from_halves(lo, hi);
}

// B 32x16: lane -> col n=lane%16; elems 0..7 = K(base..base+7), 8..15 = K(16+base..),
//          base = lane<16 ? 0 : 8
__device__ __forceinline__ v16bf load_frag_b(const __bf16* row, int k0, int lane) {
  const __bf16* p = row + k0 + ((lane < 16) ? 0 : 8);
  return frag_from_halves(*(const v8bf*)(p), *(const v8bf*)(p + 16));
}

__device__ __forceinline__ v8f wmma_bf16(v16bf a, v16bf b, v8f c) {
  return __builtin_amdgcn_wmma_f32_16x16x32_bf16(false, a, false, b, (short)0, c,
                                                 false, false);
}

// --- TDM: stage one contiguous [16][1024] bf16 tile into LDS ---------------
// D# per ISA 8.3-8.6: 2D tile (x=1024, y=16), data_size=2B, stride=1024,
// pad_enable, pad_interval=256 dwords (1024B), pad_amount=4 dwords (16B).
__device__ __forceinline__ void tdm_load_xtile(const __bf16* gsrc, char* sx_generic) {
  unsigned lds_base = (unsigned)(unsigned long long)(
      (__attribute__((address_space(3))) char*)sx_generic);
  unsigned long long ga = (unsigned long long)(const void*)gsrc;

  u32x4 g0;
  g0[0] = 1u;                                   // count=1, user descriptor
  g0[1] = lds_base;                             // lds_addr
  g0[2] = (unsigned)(ga & 0xffffffffu);         // global_addr[31:0]
  g0[3] = (unsigned)((ga >> 32) & 0x1ffffffu)   // global_addr[56:32]
          | (2u << 30);                         // type=2 ("image")

  i32x8 g1;
  g1[0] = (int)((1u << 16)                      // data_size = 2 bytes
                | (1u << 20)                    // pad_enable
                | (7u << 22)                    // pad_interval: 256 dwords
                | (3u << 25));                  // pad_amount:   4 dwords
  g1[1] = (int)((DIM & 0xffffu) << 16);         // tensor_dim0[15:0]
  g1[2] = (int)((DIM >> 16) | ((32768u & 0xffffu) << 16));  // dim0 hi | tensor_dim1 lo
  g1[3] = (int)((32768u >> 16) | ((unsigned)DIM << 16));    // dim1 hi | tile_dim0
  g1[4] = 16;                                   // tile_dim1=16, tile_dim2=0
  g1[5] = DIM;                                  // tensor_dim0_stride[31:0]
  g1[6] = 0;
  g1[7] = 0;

  i32x4 gz4 = {0, 0, 0, 0};
  i32x8 gz8 = {0, 0, 0, 0, 0, 0, 0, 0};
  // clang-23 / amdgpu-toolchain form: (g0, g1, g2, g3, g4, cpol)
  __builtin_amdgcn_tensor_load_to_lds(g0, g1, gz4, gz4, gz8, 0);
}

// --- init: state buffers, barrier, t=0 outputs -----------------------------
__global__ void k_init(float* hfp, unsigned short* hbf_u, int* bar,
                       float* out_logh, float* out_signh) {
  int i = blockIdx.x * blockDim.x + threadIdx.x;
  if (i == 0) { bar[0] = 0; bar[1] = 0; }
  if (i < 2 * BATCH * DIM) {
    hfp[i] = 0.f;
    ((__bf16*)hbf_u)[i] = (__bf16)0.f;
  }
  if (i < BATCH * DIM) {
    out_logh[i]  = -1e4f;   // LOG_ZERO
    out_signh[i] = 1.f;
  }
}

__global__ void k_f32_to_bf16(const float* __restrict__ src,
                              unsigned short* __restrict__ dst, size_t n) {
  size_t i  = (size_t)blockIdx.x * blockDim.x + threadIdx.x;
  size_t st = (size_t)gridDim.x * blockDim.x;
  __bf16* d = (__bf16*)dst;
  for (; i < n; i += st) d[i] = (__bf16)src[i];
}

// pack [Wx ; Wd] -> concat bf16 weight [2048][1024]
__global__ void k_pack_w(const float* __restrict__ Wx, const float* __restrict__ Wd,
                         unsigned short* __restrict__ w2) {
  size_t i  = (size_t)blockIdx.x * blockDim.x + threadIdx.x;
  size_t st = (size_t)gridDim.x * blockDim.x;
  __bf16* d = (__bf16*)w2;
  const size_t half = (size_t)DIM * DIM;
  const size_t n = 2 * half;
  for (; i < n; i += st) {
    float v = (i < half) ? Wx[i] : Wd[i - half];
    d[i] = (__bf16)v;
  }
}

// --- Phase 1: A = X@Wx^T + b ; G = X@Wd^T + b_delta ------------------------
// grid = 4 * (T*B/16) blocks of 256 threads; block = (mtile, N-quarter),
// wave owns 4 N-tiles. x-tile comes from LDS (TDM), B double-buffered.
__global__ void __launch_bounds__(256) k_gemm_precompute(
    const unsigned short* __restrict__ xbf_u, const unsigned short* __restrict__ w2_u,
    const float* __restrict__ b, const float* __restrict__ b_delta,
    float* __restrict__ Abuf, float* __restrict__ Gbuf) {
  const __bf16* xbf = (const __bf16*)xbf_u;
  const __bf16* w2  = (const __bf16*)w2_u;
  const int mtile = blockIdx.x >> 2;
  const int quart = blockIdx.x & 3;
  const int wave  = threadIdx.x >> 5;
  const int lane  = threadIdx.x & 31;
  const int l15   = lane & 15;
  const int mhi   = (lane < 16) ? 0 : 8;
  const int ntb   = quart * 32 + wave * 4;   // N-tile base over concat N=2048

  __shared__ __align__(16) char sx[XTILE_LDS_BYTES];

  if (threadIdx.x < 32) {                    // wave 0 drives the TDM
    tdm_load_xtile(xbf + (size_t)mtile * 16 * DIM, sx);
    __builtin_amdgcn_s_wait_tensorcnt(0);
  }
  __syncthreads();

  const __bf16* brow[4];
#pragma unroll
  for (int j = 0; j < 4; ++j)
    brow[j] = w2 + (size_t)((ntb + j) * 16 + l15) * DIM;

  v8f acc[4];
#pragma unroll
  for (int j = 0; j < 4; ++j)
    acc[j] = (v8f){0.f, 0.f, 0.f, 0.f, 0.f, 0.f, 0.f, 0.f};

  // software-pipelined K loop (unroll x2, ping-pong B register sets)
  v16bf bA[4], bB[4];
#pragma unroll
  for (int j = 0; j < 4; ++j) bA[j] = load_frag_b(brow[j], 0, lane);

  for (int k0 = 0; k0 < DIM; k0 += 64) {
#pragma unroll
    for (int j = 0; j < 4; ++j) bB[j] = load_frag_b(brow[j], k0 + 32, lane);
    v16bf af = load_frag_a_lds(sx, l15, k0, lane);
#pragma unroll
    for (int j = 0; j < 4; ++j) acc[j] = wmma_bf16(af, bA[j], acc[j]);

    const int kn = (k0 + 64 < DIM) ? (k0 + 64) : 0;   // clamp (dummy last reload)
#pragma unroll
    for (int j = 0; j < 4; ++j) bA[j] = load_frag_b(brow[j], kn, lane);
    af = load_frag_a_lds(sx, l15, k0 + 32, lane);
#pragma unroll
    for (int j = 0; j < 4; ++j) acc[j] = wmma_bf16(af, bB[j], acc[j]);
  }

#pragma unroll
  for (int j = 0; j < 4; ++j) {
    int ng = ntb + j;                        // 0..127 over concat weight
    bool isA = (ng < 64);
    int col = (ng & 63) * 16 + l15;
    float bias = isA ? b[col] : b_delta[col];
    float* dst = isA ? Abuf : Gbuf;
#pragma unroll
    for (int r = 0; r < 8; ++r) {
      int mrow = mtile * 16 + r + mhi;
      dst[(size_t)mrow * DIM + col] = acc[j][r] + bias;
    }
  }
}

// --- grid-wide sense barrier (L2 atomics) ----------------------------------
__device__ __forceinline__ void grid_barrier(int* bar) {
  __syncthreads();
  if (threadIdx.x == 0) {
    __threadfence();
    int gen = __hip_atomic_load(&bar[1], __ATOMIC_ACQUIRE, __HIP_MEMORY_SCOPE_AGENT);
    int arrived =
        __hip_atomic_fetch_add(&bar[0], 1, __ATOMIC_ACQ_REL, __HIP_MEMORY_SCOPE_AGENT);
    if (arrived == (int)gridDim.x - 1) {
      __hip_atomic_store(&bar[0], 0, __ATOMIC_RELAXED, __HIP_MEMORY_SCOPE_AGENT);
      __hip_atomic_fetch_add(&bar[1], 1, __ATOMIC_RELEASE, __HIP_MEMORY_SCOPE_AGENT);
    } else {
      while (__hip_atomic_load(&bar[1], __ATOMIC_ACQUIRE, __HIP_MEMORY_SCOPE_AGENT) ==
             gen)
        __builtin_amdgcn_s_sleep(1);
    }
  }
  __syncthreads();
}

// --- Phase 2: persistent recurrence ----------------------------------------
// 64 blocks x 256 threads. Block nt owns channels [16nt,16nt+16).
// Wave w owns K slice [128w,128w+128): R_h fragments pinned in VGPRs.
__global__ void __launch_bounds__(256) k_recurrent(
    const float* __restrict__ x, const float* __restrict__ Abuf,
    const float* __restrict__ Gbuf, const unsigned short* __restrict__ rh_u,
    const float* __restrict__ b_gate, float* __restrict__ out_output,
    float* __restrict__ out_logh, float* __restrict__ out_signh,
    float* __restrict__ out_hlin, float* __restrict__ hfp,
    unsigned short* __restrict__ hbf_u, int* __restrict__ bar) {
  const __bf16* rh = (const __bf16*)rh_u;
  __bf16* hbf = (__bf16*)hbf_u;
  const int nt    = blockIdx.x;
  const int wave  = threadIdx.x >> 5;
  const int lane  = threadIdx.x & 31;
  const int l15   = lane & 15;
  const int kbase = wave * 128;

  __shared__ float partial[8][BATCH * 16];

  // persistent B fragments: R_h rows (channels) x this wave's K slice
  v16bf bfrag[4];
  {
    const __bf16* rrow = rh + (size_t)(nt * 16 + l15) * DIM;
#pragma unroll
    for (int j = 0; j < 4; ++j) bfrag[j] = load_frag_b(rrow, kbase + j * 32, lane);
  }

  const int elem = threadIdx.x;   // C/D layout: elem = m*16 + n
  const int em = elem >> 4;       // batch row
  const int en = elem & 15;       // channel in tile
  const int ch = nt * 16 + en;
  const float bg = b_gate[ch];
  const int hidx = em * DIM + ch;

  for (int t = 0; t < T_STEPS; ++t) {
    const size_t oidx = ((size_t)t * BATCH + em) * DIM + ch;
    if (t + 1 < T_STEPS) {  // gfx1250 global_prefetch_b8 for next epilogue
      const size_t nidx = oidx + (size_t)BATCH * DIM;
      __builtin_prefetch(&Abuf[nidx], 0, 1);
      __builtin_prefetch(&Gbuf[nidx], 0, 1);
      __builtin_prefetch(&x[nidx], 0, 1);
    }

    // GEMM partial: h_prev (bf16 ping-pong) @ R_h^T over this wave's K slice
    const __bf16* hrow = hbf + (size_t)(t & 1) * BATCH * DIM + (size_t)l15 * DIM;
    v8f acc = (v8f){0.f, 0.f, 0.f, 0.f, 0.f, 0.f, 0.f, 0.f};
#pragma unroll
    for (int j = 0; j < 4; ++j)
      acc = wmma_bf16(load_frag_a(hrow, kbase + j * 32, lane), bfrag[j], acc);

#pragma unroll
    for (int r = 0; r < 8; ++r) {
      int m = r + ((lane < 16) ? 0 : 8);
      partial[wave][m * 16 + l15] = acc[r];
    }
    __syncthreads();

    float v = 0.f;
#pragma unroll
    for (int w = 0; w < 8; ++w) v += partial[w][elem];

    // fused epilogue (one element per thread)
    float a     = Abuf[oidx];
    float gz    = Gbuf[oidx];
    float hp    = hfp[(size_t)(t & 1) * BATCH * DIM + hidx];
    float cand  = tanhf(v + a);
    float delta = 1.f / (1.f + __expf(-gz));
    float hn    = (1.f - delta) * hp + delta * cand;

    float lg = __logf(fabsf(hn) + 1e-12f);
    float sg = (hn >= 0.f) ? 1.f : -1.f;
    float xt = x[oidx];
    float u  = hn + xt + bg;
    float ov = hn * (u / (1.f + __expf(-u)));

    out_hlin[oidx]   = hn;
    out_output[oidx] = ov;
    size_t lidx = ((size_t)(t + 1) * BATCH + em) * DIM + ch;
    out_logh[lidx]  = lg;
    out_signh[lidx] = sg;

    hfp[(size_t)((t + 1) & 1) * BATCH * DIM + hidx] = hn;         // fp32 carried state
    hbf[(size_t)((t + 1) & 1) * BATCH * DIM + hidx] = (__bf16)hn; // WMMA operand copy

    __threadfence();
    grid_barrier(bar);   // also covers LDS WAR for next step
  }
}

extern "C" void kernel_launch(void* const* d_in, const int* in_sizes, int n_in,
                              void* d_out, int out_size, void* d_ws, size_t ws_size,
                              hipStream_t stream) {
  (void)in_sizes; (void)n_in; (void)out_size; (void)ws_size;
  const float* x  = (const float*)d_in[0];
  const float* Wx = (const float*)d_in[1];
  const float* Rh = (const float*)d_in[2];
  const float* Wd = (const float*)d_in[3];
  const float* b  = (const float*)d_in[4];
  const float* bd = (const float*)d_in[5];
  const float* bg = (const float*)d_in[6];

  const size_t TBD  = (size_t)T_STEPS * BATCH * DIM;
  const size_t T1BD = (size_t)(T_STEPS + 1) * BATCH * DIM;
  float* out        = (float*)d_out;
  float* out_output = out;
  float* out_logh   = out + TBD;
  float* out_signh  = out + TBD + T1BD;
  float* out_hlin   = out + TBD + 2 * T1BD;

  char* ws = (char*)d_ws;
  size_t off = 0;
  int* bar             = (int*)(ws + off);            off += 256;
  unsigned short* xbf  = (unsigned short*)(ws + off); off += TBD * 2;
  unsigned short* w2   = (unsigned short*)(ws + off); off += (size_t)2 * DIM * DIM * 2;
  unsigned short* rhb  = (unsigned short*)(ws + off); off += (size_t)DIM * DIM * 2;
  float* Abuf          = (float*)(ws + off);          off += TBD * 4;
  float* Gbuf          = (float*)(ws + off);          off += TBD * 4;
  float* hfp           = (float*)(ws + off);          off += (size_t)2 * BATCH * DIM * 4;
  unsigned short* hbf  = (unsigned short*)(ws + off); off += (size_t)2 * BATCH * DIM * 2;

  k_init<<<(2 * BATCH * DIM + 255) / 256, 256, 0, stream>>>(hfp, hbf, bar,
                                                            out_logh, out_signh);
  k_f32_to_bf16<<<4096, 256, 0, stream>>>(x, xbf, TBD);
  k_pack_w<<<2048, 256, 0, stream>>>(Wx, Wd, w2);
  k_f32_to_bf16<<<1024, 256, 0, stream>>>(Rh, rhb, (size_t)DIM * DIM);
  k_gemm_precompute<<<(T_STEPS * BATCH / 16) * 4, 256, 0, stream>>>(xbf, w2, b, bd,
                                                                    Abuf, Gbuf);
  k_recurrent<<<64, 256, 0, stream>>>(x, Abuf, Gbuf, rhb, bg, out_output, out_logh,
                                      out_signh, out_hlin, hfp, hbf, bar);
}
